// MambaTokenMixer_18777597018416
// MI455X (gfx1250) — matrix operand order repaired and checked by reading, not compiled, not run on hardware
//
#include <hip/hip_runtime.h>
#include <hip/hip_bf16.h>
#include <math.h>

// ---------------------------------------------------------------------------
// MI455X (gfx1250) Mamba token mixer.
//   B=4, C=256, H=W=64 -> L=4096, M=B*L=16384, D_INNER=512, D_STATE=16,
//   DT_RANK=16, xproj N=48, conv K=4.
// GEMMs: v_wmma_f32_16x16x32_f16. Operands are pre-converted to f16 in global
// memory (weights once, activations by their producers), so unguarded GEMMs
// stage LDS tiles with global_load_async_to_lds_b128 (ASYNCcnt) in a
// double-buffered single-barrier pipeline. Ragged GEMMs (N=48 / K=16) use a
// synchronous register-staged double buffer with zero-fill.
// Selective scan is chunk-parallel (32 chunks x 128 steps).
// ---------------------------------------------------------------------------

typedef __attribute__((ext_vector_type(16))) _Float16 v16h;
typedef __attribute__((ext_vector_type(8)))  _Float16 v8h;
typedef __attribute__((ext_vector_type(8)))  float    v8f;

#define BATCH   4
#define CDIM    256
#define HWDIM   4096
#define MROWS   (BATCH * HWDIM)   // 16384
#define DIN     512
#define DST     16
#define XPN     48                // DT_RANK + 2*DST
#define NCHUNK  32
#define CHUNK   128               // HWDIM / NCHUNK
#define LDSTR   40                // LDS row stride in halves (80B, 16B-aligned)

constexpr int EPI_NONE = 0;  // store acc
constexpr int EPI_GATE = 1;  // sigmoid(silu(acc + bias))
constexpr int EPI_SP   = 2;  // softplus(acc + bias)
constexpr int EPI_MUL  = 3;  // acc * aux[m,n]
constexpr int EPI_ADD  = 4;  // acc + aux[m,n]

__device__ __forceinline__ float sigmoidf_(float v) { return 1.f / (1.f + __expf(-v)); }

// ---------------------------------------------------------------------------
// f32 -> f16 conversion (weights, once per call)
// ---------------------------------------------------------------------------
__global__ void __launch_bounds__(256)
cvt_h_kernel(const float* __restrict__ src, _Float16* __restrict__ dst, int n) {
  const int i = blockIdx.x * 256 + threadIdx.x;
  if (i < n) dst[i] = (_Float16)src[i];
}

// ---------------------------------------------------------------------------
// NCHW(b,c,l) -> (b*L+l, c) transpose, writes f16 (only a GEMM consumes it)
// ---------------------------------------------------------------------------
__global__ void __launch_bounds__(256)
permute_in_kernel(const float* __restrict__ x, _Float16* __restrict__ xph) {
  __shared__ float tile[32][33];
  const int b  = blockIdx.z;
  const int l0 = blockIdx.x * 32;
  const int c0 = blockIdx.y * 32;
  const int tx = threadIdx.x, ty = threadIdx.y;   // 32 x 8
  #pragma unroll
  for (int i = 0; i < 32; i += 8)
    tile[ty + i][tx] = x[(size_t)(b * CDIM + c0 + ty + i) * HWDIM + l0 + tx];
  __syncthreads();
  #pragma unroll
  for (int i = 0; i < 32; i += 8)
    xph[(size_t)(b * HWDIM + l0 + ty + i) * CDIM + c0 + tx] = (_Float16)tile[tx][ty + i];
}

// out[b,c,l] = x[b,c,l] + sum[b*L+l, c]   (inverse transpose + residual)
__global__ void __launch_bounds__(256)
final_out_kernel(const float* __restrict__ x, const float* __restrict__ summ,
                 float* __restrict__ out) {
  __shared__ float tile[32][33];
  const int b  = blockIdx.z;
  const int l0 = blockIdx.x * 32;
  const int c0 = blockIdx.y * 32;
  const int tx = threadIdx.x, ty = threadIdx.y;
  #pragma unroll
  for (int i = 0; i < 32; i += 8)
    tile[ty + i][tx] = summ[(size_t)(b * HWDIM + l0 + ty + i) * CDIM + c0 + tx];
  __syncthreads();
  #pragma unroll
  for (int i = 0; i < 32; i += 8) {
    const size_t xi = (size_t)(b * CDIM + c0 + ty + i) * HWDIM + l0 + tx;
    out[xi] = x[xi] + tile[tx][ty + i];
  }
}

// ---------------------------------------------------------------------------
// WMMA f16 GEMM:  O[M,N] (+epi) = A[M,K](f16) @ W[N,K]^T(f16)
// Block tile 64x64, BK=32, 8 waves; each wave -> 16x32 (two 16x16 accs).
// LDS: As[m][k] / Bs[n][k], row stride 40 halves (16B-aligned, b128 frag
// reads hit distinct banks: 20*m mod 64 is a permutation of 16 values).
// Fragments as used here (wave32; A and B share contiguous k, not the ISA table order):
//   A 16x32 : lane(g,mi): half j -> K = 16g + j                 (two v8h)
//   B 32x16 : lane(g,n):  half j -> k = 16g + j                 (two v8h)
//   C/D     : vgpr i -> M = i + 8g, N = lane&15
// !GUARD (M%64==0, N%64==0, K%32==0): tiles staged via
//   global_load_async_to_lds_b128, double-buffered, one barrier per K-step.
// GUARD: synchronous v8h loads with all-or-nothing zero-fill per 8-half
//   chunk (covers N=48 and K=16 exactly).
// ---------------------------------------------------------------------------
template <int EPI, bool GUARD, bool STF32, bool STF16>
__global__ void __launch_bounds__(256)
gemm_h(const _Float16* __restrict__ A, int lda,
       const _Float16* __restrict__ W, int ldw,
       float* __restrict__ O, _Float16* __restrict__ Oh, int ldo,
       const float* __restrict__ bias,
       const float* __restrict__ aux, int ldaux,
       int M, int N, int K) {
  __shared__ __align__(16) _Float16 As[2][64][LDSTR];
  __shared__ __align__(16) _Float16 Bs[2][64][LDSTR];

  const int tid  = threadIdx.x;
  const int m0   = blockIdx.y * 64;
  const int n0   = blockIdx.x * 64;
  const int w    = tid >> 5;
  const int lane = tid & 31;
  const int g    = lane >> 4;
  const int mi   = lane & 15;
  const int wm   = (w & 3) * 16;
  const int wn   = (w >> 2) * 32;

  v8f acc0 = {}; v8f acc1 = {};

  const int ldr = tid >> 2;          // 0..63 tile row
  const int ldc = (tid & 3) * 8;     // 0,8,16,24 (16B-aligned k base)
  const size_t aoff = (size_t)(m0 + ldr) * lda + ldc;
  const size_t boff = (size_t)(n0 + ldr) * ldw + ldc;
  const v8h zh = {};

  int p = 0;
  if (!GUARD) {
    // LDS destination addresses (low 32 bits of flat shared ptr = LDS offset)
    const unsigned la0 = (unsigned)(size_t)&As[0][ldr][ldc];
    const unsigned la1 = (unsigned)(size_t)&As[1][ldr][ldc];
    const unsigned lb0 = (unsigned)(size_t)&Bs[0][ldr][ldc];
    const unsigned lb1 = (unsigned)(size_t)&Bs[1][ldr][ldc];
    auto issue = [&](int buf, int k0) {
      const unsigned long long ga = (unsigned long long)(size_t)(A + aoff + k0);
      const unsigned long long gb = (unsigned long long)(size_t)(W + boff + k0);
      asm volatile("global_load_async_to_lds_b128 %0, %1, off"
                   :: "v"(buf ? la1 : la0), "v"(ga) : "memory");
      asm volatile("global_load_async_to_lds_b128 %0, %1, off"
                   :: "v"(buf ? lb1 : lb0), "v"(gb) : "memory");
    };
    issue(0, 0);
    asm volatile("s_wait_asynccnt 0x0" ::: "memory");
    for (int k0 = 0; k0 < K; k0 += 32) {
      __syncthreads();                       // buf p fully written, p^1 free
      const bool more = (k0 + 32) < K;
      if (more) issue(p ^ 1, k0 + 32);       // async fill of other buffer

      const _Float16* arow = &As[p][wm + mi][0];
      const v16h af = __builtin_shufflevector(
          *(const v8h*)(arow + 16 * g), *(const v8h*)(arow + 16 * g + 8),
          0, 1, 2, 3, 4, 5, 6, 7, 8, 9, 10, 11, 12, 13, 14, 15);
      const _Float16* brow0 = &Bs[p][wn + mi][16 * g];
      const v16h bf0 = __builtin_shufflevector(
          *(const v8h*)brow0, *(const v8h*)(brow0 + 8),
          0, 1, 2, 3, 4, 5, 6, 7, 8, 9, 10, 11, 12, 13, 14, 15);
      const _Float16* brow1 = &Bs[p][wn + 16 + mi][16 * g];
      const v16h bf1 = __builtin_shufflevector(
          *(const v8h*)brow1, *(const v8h*)(brow1 + 8),
          0, 1, 2, 3, 4, 5, 6, 7, 8, 9, 10, 11, 12, 13, 14, 15);

      acc0 = __builtin_amdgcn_wmma_f32_16x16x32_f16(false, af, false, bf0,
                                                    (short)0, acc0, false, false);
      acc1 = __builtin_amdgcn_wmma_f32_16x16x32_f16(false, af, false, bf1,
                                                    (short)0, acc1, false, false);
      if (more) asm volatile("s_wait_asynccnt 0x0" ::: "memory");
      p ^= 1;
    }
  } else {
    const bool rokA = (m0 + ldr) < M;
    const bool rokB = (n0 + ldr) < N;
    v8h ra = (rokA && ldc < K) ? *(const v8h*)(A + aoff) : zh;
    v8h rb = (rokB && ldc < K) ? *(const v8h*)(W + boff) : zh;
    *(v8h*)&As[0][ldr][ldc] = ra;
    *(v8h*)&Bs[0][ldr][ldc] = rb;
    for (int k0 = 0; k0 < K; k0 += 32) {
      __syncthreads();
      const bool more = (k0 + 32) < K;
      v8h na = zh, nb = zh;
      if (more) {
        na = (rokA && (k0 + 32 + ldc) < K) ? *(const v8h*)(A + aoff + k0 + 32) : zh;
        nb = (rokB && (k0 + 32 + ldc) < K) ? *(const v8h*)(W + boff + k0 + 32) : zh;
      }
      const _Float16* arow = &As[p][wm + mi][0];
      const v16h af = __builtin_shufflevector(
          *(const v8h*)(arow + 16 * g), *(const v8h*)(arow + 16 * g + 8),
          0, 1, 2, 3, 4, 5, 6, 7, 8, 9, 10, 11, 12, 13, 14, 15);
      const _Float16* brow0 = &Bs[p][wn + mi][16 * g];
      const v16h bf0 = __builtin_shufflevector(
          *(const v8h*)brow0, *(const v8h*)(brow0 + 8),
          0, 1, 2, 3, 4, 5, 6, 7, 8, 9, 10, 11, 12, 13, 14, 15);
      const _Float16* brow1 = &Bs[p][wn + 16 + mi][16 * g];
      const v16h bf1 = __builtin_shufflevector(
          *(const v8h*)brow1, *(const v8h*)(brow1 + 8),
          0, 1, 2, 3, 4, 5, 6, 7, 8, 9, 10, 11, 12, 13, 14, 15);

      acc0 = __builtin_amdgcn_wmma_f32_16x16x32_f16(false, af, false, bf0,
                                                    (short)0, acc0, false, false);
      acc1 = __builtin_amdgcn_wmma_f32_16x16x32_f16(false, af, false, bf1,
                                                    (short)0, acc1, false, false);
      if (more) {
        *(v8h*)&As[p ^ 1][ldr][ldc] = na;
        *(v8h*)&Bs[p ^ 1][ldr][ldc] = nb;
      }
      p ^= 1;
    }
  }

  // ---- epilogue ----
  #pragma unroll
  for (int i = 0; i < 8; i++) {
    const int mrow = m0 + wm + i + 8 * g;
    #pragma unroll
    for (int t = 0; t < 2; t++) {
      const int ncol = n0 + wn + t * 16 + mi;
      if (GUARD && ncol >= N) continue;
      float v = (t == 0) ? acc0[i] : acc1[i];
      if (EPI == EPI_GATE) {
        v += bias[ncol];
        const float s = v * sigmoidf_(v);
        v = sigmoidf_(s);
      } else if (EPI == EPI_SP) {
        v += bias[ncol];
        v = fmaxf(v, 0.f) + log1pf(__expf(-fabsf(v)));
      } else if (EPI == EPI_MUL) {
        v *= aux[(size_t)mrow * ldaux + ncol];
      } else if (EPI == EPI_ADD) {
        v += aux[(size_t)mrow * ldaux + ncol];
      }
      if (STF32) O[(size_t)mrow * ldo + ncol] = v;
      if (STF16) Oh[(size_t)mrow * ldo + ncol] = (_Float16)v;
    }
  }
}

// ---------------------------------------------------------------------------
// LayerNorm over C=256: one wave per token row; output f16 (GEMM-only use).
// ---------------------------------------------------------------------------
__global__ void __launch_bounds__(256)
ln_kernel(const float* __restrict__ tok, const float* __restrict__ gam,
          const float* __restrict__ bet, _Float16* __restrict__ outh) {
  const int wave = threadIdx.x >> 5, lane = threadIdx.x & 31;
  const int m = blockIdx.x * 8 + wave;
  const float* row = tok + (size_t)m * CDIM;
  float v[8], s = 0.f, s2 = 0.f;
  const float4 p0 = *(const float4*)(row + lane * 8);
  const float4 p1 = *(const float4*)(row + lane * 8 + 4);
  v[0] = p0.x; v[1] = p0.y; v[2] = p0.z; v[3] = p0.w;
  v[4] = p1.x; v[5] = p1.y; v[6] = p1.z; v[7] = p1.w;
  #pragma unroll
  for (int i = 0; i < 8; i++) { s += v[i]; s2 += v[i] * v[i]; }
  #pragma unroll
  for (int off = 16; off >= 1; off >>= 1) {
    s  += __shfl_xor(s,  off, 32);
    s2 += __shfl_xor(s2, off, 32);
  }
  const float mu = s * (1.f / CDIM);
  const float var = s2 * (1.f / CDIM) - mu * mu;
  const float r = rsqrtf(var + 1e-5f);
  _Float16* orow = outh + (size_t)m * CDIM;
  #pragma unroll
  for (int i = 0; i < 8; i++) {
    const int c = lane * 8 + i;
    orow[c] = (_Float16)((v[i] - mu) * r * gam[c] + bet[c]);
  }
}

// ---------------------------------------------------------------------------
// Depthwise causal conv (K=4) + bias + SiLU; writes f32 (scan) + f16 (GEMM).
// ---------------------------------------------------------------------------
__global__ void __launch_bounds__(256)
conv_silu_kernel(const float* __restrict__ xz, const float* __restrict__ cw,
                 const float* __restrict__ cb, float* __restrict__ xmc,
                 _Float16* __restrict__ xmch) {
  const int idx = blockIdx.x * 256 + threadIdx.x;   // M*512 threads
  const int d = idx & (DIN - 1);
  const int m = idx >> 9;
  const int l = m & (HWDIM - 1);
  float acc = cb[d];
  #pragma unroll
  for (int k = 0; k < 4; k++) {
    const int ls = l + k - 3;
    if (ls >= 0) acc += xz[(size_t)(m + k - 3) * (2 * DIN) + d] * cw[d * 4 + k];
  }
  const float r = acc * sigmoidf_(acc);
  xmc[(size_t)m * DIN + d]  = r;
  xmch[(size_t)m * DIN + d] = (_Float16)r;
}

// ---------------------------------------------------------------------------
// Selective scan, pass A: chunk-local scan (h0=0); y_local, final h, S=sum(dt)
// grid = (B*NCHUNK, 2), block 256 -> one thread per (b,chunk,d).
// ---------------------------------------------------------------------------
__global__ void __launch_bounds__(256)
scan_local_kernel(const float* __restrict__ dt, const float* __restrict__ xmc,
                  const float* __restrict__ dbl, const float* __restrict__ Alog,
                  float* __restrict__ ybuf, float* __restrict__ hend,
                  float* __restrict__ Ssum) {
  const int d  = blockIdx.y * 256 + threadIdx.x;
  const int bc = blockIdx.x;
  const int b  = bc >> 5, c = bc & (NCHUNK - 1);
  float Ad[DST], h[DST];
  #pragma unroll
  for (int n = 0; n < DST; n++) { Ad[n] = -__expf(Alog[d * DST + n]); h[n] = 0.f; }
  float S = 0.f;
  const int mbase = b * HWDIM + c * CHUNK;
  for (int t = 0; t < CHUNK; t++) {
    const int m = mbase + t;
    const float dtv = dt[(size_t)m * DIN + d];
    const float xv  = xmc[(size_t)m * DIN + d];
    __builtin_prefetch(dt  + (size_t)(m + 8) * DIN + d, 0, 1);
    __builtin_prefetch(xmc + (size_t)(m + 8) * DIN + d, 0, 1);
    S += dtv;
    const float dtx = dtv * xv;
    const float* bcrow = dbl + (size_t)m * XPN;   // uniform across block
    float y = 0.f;
    #pragma unroll
    for (int n = 0; n < DST; n++) {
      const float dA = __expf(dtv * Ad[n]);
      h[n] = dA * h[n] + dtx * bcrow[DST + n];
      y += h[n] * bcrow[2 * DST + n];
    }
    ybuf[(size_t)m * DIN + d] = y;
  }
  const size_t hb = ((size_t)bc * DIN + d) * DST;
  #pragma unroll
  for (int n = 0; n < DST; n++) hend[hb + n] = h[n];
  Ssum[(size_t)bc * DIN + d] = S;
}

// Pass B: sequential carry over 32 chunk summaries (one thread per (b,d,n)).
__global__ void __launch_bounds__(256)
scan_carry_kernel(const float* __restrict__ hend, const float* __restrict__ Ssum,
                  const float* __restrict__ Alog, float* __restrict__ carry) {
  const int idx = blockIdx.x * 256 + threadIdx.x;
  const int n = idx & (DST - 1);
  const int d = (idx >> 4) & (DIN - 1);
  const int b = idx >> 13;
  const float An = -__expf(Alog[d * DST + n]);
  float cy = 0.f;
  for (int c = 0; c < NCHUNK; c++) {
    const size_t base = (size_t)(b * NCHUNK + c) * DIN + d;
    carry[base * DST + n] = cy;
    cy = __expf(An * Ssum[base]) * cy + hend[base * DST + n];
  }
}

// Pass C: y += C_t . (exp(A*cumsum(dt)) * carry); +D*x; *silu(z) -> ym (f16)
__global__ void __launch_bounds__(256)
scan_fix_gate_kernel(const float* __restrict__ dt, const float* __restrict__ xmc,
                     const float* __restrict__ dbl, const float* __restrict__ Alog,
                     const float* __restrict__ carry, const float* __restrict__ Dp,
                     const float* __restrict__ xz, const float* __restrict__ ybuf,
                     _Float16* __restrict__ ymh) {
  const int d  = blockIdx.y * 256 + threadIdx.x;
  const int bc = blockIdx.x;
  const int b  = bc >> 5, c = bc & (NCHUNK - 1);
  const bool hasC = (c != 0);
  float Ad[DST], cr[DST];
  const size_t cbase = ((size_t)bc * DIN + d) * DST;
  #pragma unroll
  for (int n = 0; n < DST; n++) {
    Ad[n] = -__expf(Alog[d * DST + n]);
    cr[n] = carry[cbase + n];
  }
  const float Dd = Dp[d];
  float cum = 0.f;
  const int mbase = b * HWDIM + c * CHUNK;
  for (int t = 0; t < CHUNK; t++) {
    const int m = mbase + t;
    const float dtv = dt[(size_t)m * DIN + d];
    cum += dtv;
    float yfix = 0.f;
    if (hasC) {
      const float* crow = dbl + (size_t)m * XPN + 2 * DST;  // uniform
      #pragma unroll
      for (int n = 0; n < DST; n++)
        yfix += crow[n] * __expf(Ad[n] * cum) * cr[n];
    }
    const float y = ybuf[(size_t)m * DIN + d] + yfix + Dd * xmc[(size_t)m * DIN + d];
    const float z = xz[(size_t)m * (2 * DIN) + DIN + d];
    ymh[(size_t)m * DIN + d] = (_Float16)(y * (z * sigmoidf_(z)));
  }
}

// ---------------------------------------------------------------------------
// Orchestration. f32 region ~230 MB + f16 region ~70 MB of workspace.
// ---------------------------------------------------------------------------
extern "C" void kernel_launch(void* const* d_in, const int* in_sizes, int n_in,
                              void* d_out, int out_size, void* d_ws, size_t ws_size,
                              hipStream_t stream) {
  const float* x        = (const float*)d_in[0];
  const float* w_in     = (const float*)d_in[1];   // (256,256)
  const float* w_out    = (const float*)d_in[2];   // (256,256)
  const float* w_gate   = (const float*)d_in[3];   // (256,256)
  const float* b_gate   = (const float*)d_in[4];
  const float* ln_g     = (const float*)d_in[5];
  const float* ln_b     = (const float*)d_in[6];
  const float* m_in_w   = (const float*)d_in[7];   // (1024,256)
  const float* m_conv_w = (const float*)d_in[8];   // (512,4)
  const float* m_conv_b = (const float*)d_in[9];
  const float* m_xproj  = (const float*)d_in[10];  // (48,512)
  const float* m_dt_w   = (const float*)d_in[11];  // (512,16)
  const float* m_dt_b   = (const float*)d_in[12];
  const float* m_Alog   = (const float*)d_in[13];  // (512,16)
  const float* m_D      = (const float*)d_in[14];
  const float* m_out_w  = (const float*)d_in[15];  // (256,512)
  float* out = (float*)d_out;

  const size_t MC = (size_t)MROWS * CDIM;     // 4,194,304
  const size_t MD = (size_t)MROWS * DIN;      // 8,388,608

  // ---- f32 region ----
  float* f = (float*)d_ws;
  float* tokens = f;                 f += MC;
  float* zbuf   = f;                 f += MC;
  float* xz     = f;                 f += 2 * MD;
  float* xmc    = f;                 f += MD;
  float* dbl    = f;                 f += (size_t)MROWS * XPN;
  float* dtb    = f;                 f += MD;
  float* ybuf   = f;                 f += MD;
  float* hend   = f;                 f += (size_t)BATCH * NCHUNK * DIN * DST;
  float* Ss     = f;                 f += (size_t)BATCH * NCHUNK * DIN;
  float* carry  = f;                 f += (size_t)BATCH * NCHUNK * DIN * DST;
  float* summ   = f;                 f += MC;
  // ---- f16 region ----
  _Float16* h = (_Float16*)f;
  _Float16* xperm_h  = h;            h += MC;
  _Float16* tokens_h = h;            h += MC;
  _Float16* tbuf_h   = h;            h += MC;
  _Float16* xmc_h    = h;            h += MD;
  _Float16* dbl_h    = h;            h += (size_t)MROWS * XPN;
  _Float16* ym_h     = h;            h += MD;
  _Float16* gz_h     = h;            h += MC;
  _Float16* w_in_h   = h;            h += CDIM * CDIM;
  _Float16* w_gate_h = h;            h += CDIM * CDIM;
  _Float16* w_out_h  = h;            h += CDIM * CDIM;
  _Float16* m_in_h   = h;            h += 2 * DIN * CDIM;
  _Float16* xproj_h  = h;            h += XPN * DIN;
  _Float16* dtw_h    = h;            h += DIN * DST;
  _Float16* outw_h   = h;            h += CDIM * DIN;

  const dim3 pblk(32, 8);
  const dim3 pgrid(HWDIM / 32, CDIM / 32, BATCH);
  auto ggrid = [](int M, int N) { return dim3((N + 63) / 64, (M + 63) / 64); };
  auto cvt = [&](const float* s, _Float16* d, int n) {
    cvt_h_kernel<<<(n + 255) / 256, 256, 0, stream>>>(s, d, n);
  };

  // 0) weights -> f16 (once per launch; tiny)
  cvt(w_in,    w_in_h,   CDIM * CDIM);
  cvt(w_gate,  w_gate_h, CDIM * CDIM);
  cvt(w_out,   w_out_h,  CDIM * CDIM);
  cvt(m_in_w,  m_in_h,   2 * DIN * CDIM);
  cvt(m_xproj, xproj_h,  XPN * DIN);
  cvt(m_dt_w,  dtw_h,    DIN * DST);
  cvt(m_out_w, outw_h,   CDIM * DIN);

  // 1) NCHW -> (M, C) f16
  permute_in_kernel<<<pgrid, pblk, 0, stream>>>(x, xperm_h);
  // 2) tokens = xperm @ w_in^T   (f32 for LN/residual, f16 for gate GEMM)
  gemm_h<EPI_NONE, false, true, true><<<ggrid(MROWS, CDIM), 256, 0, stream>>>(
      xperm_h, CDIM, w_in_h, CDIM, tokens, tokens_h, CDIM,
      nullptr, nullptr, 0, MROWS, CDIM, CDIM);
  // 3) z = sigmoid(silu(tokens @ w_gate^T + b_gate))
  gemm_h<EPI_GATE, false, true, false><<<ggrid(MROWS, CDIM), 256, 0, stream>>>(
      tokens_h, CDIM, w_gate_h, CDIM, zbuf, nullptr, CDIM,
      b_gate, nullptr, 0, MROWS, CDIM, CDIM);
  // 4) t = LN(tokens) -> f16
  ln_kernel<<<MROWS / 8, 256, 0, stream>>>(tokens, ln_g, ln_b, tbuf_h);
  // 5) xz = t @ m_in_w^T   (N=1024)
  gemm_h<EPI_NONE, false, true, false><<<ggrid(MROWS, 2 * DIN), 256, 0, stream>>>(
      tbuf_h, CDIM, m_in_h, CDIM, xz, nullptr, 2 * DIN,
      nullptr, nullptr, 0, MROWS, 2 * DIN, CDIM);
  // 6) xmc = silu(causal_conv(xm) + b)  (f32 + f16)
  conv_silu_kernel<<<(MROWS * DIN) / 256, 256, 0, stream>>>(
      xz, m_conv_w, m_conv_b, xmc, xmc_h);
  // 7) dbl = xmc @ xproj^T  (N=48 ragged -> GUARD; f32 + f16)
  gemm_h<EPI_NONE, true, true, true><<<ggrid(MROWS, XPN), 256, 0, stream>>>(
      xmc_h, DIN, xproj_h, DIN, dbl, dbl_h, XPN,
      nullptr, nullptr, 0, MROWS, XPN, DIN);
  // 8) dt = softplus(dbl[:, :16] @ dt_w^T + dt_b)  (K=16 ragged -> GUARD)
  gemm_h<EPI_SP, true, true, false><<<ggrid(MROWS, DIN), 256, 0, stream>>>(
      dbl_h, XPN, dtw_h, DST, dtb, nullptr, DIN,
      m_dt_b, nullptr, 0, MROWS, DIN, DST);
  // 9-11) chunked selective scan + gating
  scan_local_kernel<<<dim3(BATCH * NCHUNK, DIN / 256), 256, 0, stream>>>(
      dtb, xmc, dbl, m_Alog, ybuf, hend, Ss);
  scan_carry_kernel<<<(BATCH * DIN * DST) / 256, 256, 0, stream>>>(
      hend, Ss, m_Alog, carry);
  scan_fix_gate_kernel<<<dim3(BATCH * NCHUNK, DIN / 256), 256, 0, stream>>>(
      dtb, xmc, dbl, m_Alog, carry, m_D, xz, ybuf, ym_h);
  // 12) gz = (ym @ m_out_w^T) * z  -> f16
  gemm_h<EPI_MUL, false, false, true><<<ggrid(MROWS, CDIM), 256, 0, stream>>>(
      ym_h, DIN, outw_h, DIN, nullptr, gz_h, CDIM,
      nullptr, zbuf, CDIM, MROWS, CDIM, DIN);
  // 13) summ = tokens + gz @ w_out^T
  gemm_h<EPI_ADD, false, true, false><<<ggrid(MROWS, CDIM), 256, 0, stream>>>(
      gz_h, CDIM, w_out_h, CDIM, summ, nullptr, CDIM,
      nullptr, tokens, CDIM, MROWS, CDIM, CDIM);
  // 14) out = x + permute_back(summ)
  final_out_kernel<<<pgrid, pblk, 0, stream>>>(x, summ, out);
}
